// ConvPDDecoder_58729382805665
// MI455X (gfx1250) — matrix operand order repaired and checked by reading, not compiled, not run on hardware
//
#include <hip/hip_runtime.h>
#include <hip/hip_bf16.h>

typedef __attribute__((ext_vector_type(16))) _Float16 v16h;
typedef __attribute__((ext_vector_type(8)))  _Float16 v8h;
typedef __attribute__((ext_vector_type(8)))  float    v8f;

#define BB 4
#define CC 4
#define NN 2048
#define MM 4096
#define KK 32

// ---------------------------------------------------------------------------
// Kernel 1: zt[b,c,n,k] = sum_m exp(-0.5*(xz[m]-x[b,n])^2/s) * z[b,c,m,k]
// One block = (b, c, 128-row n-block). 8 waves, each owns a 16-row n-tile and
// both 16-col k-tiles. Reduction over M in chunks of 32 via wmma_f32_16x16x32_f16.
// A operand (w^T tile) is synthesized in registers with v_exp; B operand is z
// converted f32->f16 in the 16-bit B VGPR layout.
// ---------------------------------------------------------------------------
__global__ void __launch_bounds__(256) zt_wmma_kernel(
    const float* __restrict__ xz, const float* __restrict__ x,
    const float* __restrict__ z, const float* __restrict__ log_scale,
    _Float16* __restrict__ ztp) {
  const int blk  = blockIdx.x;        // 0..255
  const int nblk = blk & 15;
  const int c    = (blk >> 4) & 3;
  const int b    = blk >> 6;
  const int tid  = threadIdx.x;
  const int wave = tid >> 5;
  const int lane = tid & 31;
  const int l16  = lane >> 4;         // 0 = lanes 0-15, 1 = lanes 16-31
  const int lm   = lane & 15;

  const int ntile = nblk * 128 + wave * 16;
  const int nrow  = ntile + lm;       // this lane's A-row index (n)

  const float xn    = x[b * NN + nrow];             // D == 1
  const float scale = -0.5f * __expf(-log_scale[0]); // -0.5 / exp(log_scale)

  const float* __restrict__ zb = z + (size_t)((b * CC + c) * MM) * KK;

  v8f acc0 = {};
  v8f acc1 = {};

  for (int mc = 0; mc < MM; mc += 32) {
    // A operand: 16x32 tile of w^T. k order used here (same as B):
    //   kA(h,lane) = h + (lane>=16 ? 16 : 0)
    v16h A;
#pragma unroll
    for (int h = 0; h < 16; ++h) {
      const int kA = h + (l16 ? 16 : 0);
      const float d = xz[mc + kA] - xn;
      A[h] = (_Float16)__expf(scale * d * d);
    }
    // B operand: 32x16 tile of z. k order used here (same as A):
    //   row m = h + (lane>=16 ? 16 : 0), col = lane&15
    v16h B0, B1;
#pragma unroll
    for (int h = 0; h < 16; ++h) {
      const int m = mc + h + (l16 ? 16 : 0);
      const float* __restrict__ row = zb + (size_t)m * KK;
      B0[h] = (_Float16)row[lm];        // k-tile 0: cols 0..15
      B1[h] = (_Float16)row[16 + lm];   // k-tile 1: cols 16..31
    }
    acc0 = __builtin_amdgcn_wmma_f32_16x16x32_f16(false, A, false, B0,
                                                  (short)0, acc0, false, false);
    acc1 = __builtin_amdgcn_wmma_f32_16x16x32_f16(false, A, false, B1,
                                                  (short)0, acc1, false, false);
  }

  // Store zt as f16, row-major [n][32]. C/D layout: lane holds column lm,
  // rows r (+8 for lanes 16-31).
  _Float16* __restrict__ zo = ztp + (size_t)((b * CC + c) * NN) * KK;
#pragma unroll
  for (int r = 0; r < 8; ++r) {
    const int n = ntile + (l16 ? (r + 8) : r);
    zo[(size_t)n * KK + lm]      = (_Float16)acc0[r];
    zo[(size_t)n * KK + 16 + lm] = (_Float16)acc1[r];
  }
}

// ---------------------------------------------------------------------------
// Kernel 2: out[b,c,n,m'] = sum_k zt[n,k]*zt[m',k].  K = 32 -> exactly one
// wmma_f32_16x16x32_f16 per 16x16 output tile.  Block computes 128x128;
// each of 8 waves computes a 32x64 sub-block (2 A-tiles x 4 B-tiles = 8 WMMAs).
// zt row-major f16 feeds both operand layouts with b128 loads, no transpose.
// ---------------------------------------------------------------------------
__global__ void __launch_bounds__(256) gram_wmma_kernel(
    const _Float16* __restrict__ ztp, float* __restrict__ outp) {
  const int bc   = blockIdx.z;        // b*4 + c
  const int tid  = threadIdx.x;
  const int wave = tid >> 5;
  const int lane = tid & 31;
  const int l16  = lane >> 4;
  const int lm   = lane & 15;

  const int rb = blockIdx.y * 128 + (wave & 3) * 32;  // 2 row tiles
  const int cb = blockIdx.x * 128 + (wave >> 2) * 64; // 4 col tiles

  const _Float16* __restrict__ zt = ztp + (size_t)bc * NN * KK;

  // A tiles: rows rb + a*16.  contiguous k order (as B), per lane, halves at
  // k = l16*16 + {0..15}  -> two 16B loads.
  v16h A[2];
#pragma unroll
  for (int a = 0; a < 2; ++a) {
    const _Float16* p = zt + (size_t)(rb + a * 16 + lm) * KK + l16 * 16;
    const v8h lo = *(const v8h*)p;
    const v8h hi = *(const v8h*)(p + 8);
#pragma unroll
    for (int h = 0; h < 8; ++h) { A[a][h] = lo[h]; A[a][h + 8] = hi[h]; }
  }

  // B tiles: cols cb + j*16 (B is zt^T).  contiguous k order (as A), per lane, halves at
  // k = l16*16 + {0..15} of row (cb + j*16 + lm)  -> one contiguous 32B run.
  v16h Bt[4];
#pragma unroll
  for (int j = 0; j < 4; ++j) {
    const _Float16* p = zt + (size_t)(cb + j * 16 + lm) * KK + l16 * 16;
    const v8h lo = *(const v8h*)p;
    const v8h hi = *(const v8h*)(p + 8);
#pragma unroll
    for (int h = 0; h < 8; ++h) { Bt[j][h] = lo[h]; Bt[j][h + 8] = hi[h]; }
  }

  v8f acc[2][4];
#pragma unroll
  for (int a = 0; a < 2; ++a)
#pragma unroll
    for (int j = 0; j < 4; ++j) acc[a][j] = (v8f){};

#pragma unroll
  for (int a = 0; a < 2; ++a)
#pragma unroll
    for (int j = 0; j < 4; ++j)
      acc[a][j] = __builtin_amdgcn_wmma_f32_16x16x32_f16(
          false, A[a], false, Bt[j], (short)0, acc[a][j], false, false);

  // Store: out tensor starts after the 4096-float xz copy (tuple element 0).
  float* __restrict__ ob = outp + (size_t)bc * NN * NN;
#pragma unroll
  for (int a = 0; a < 2; ++a) {
#pragma unroll
    for (int j = 0; j < 4; ++j) {
      const int mcol = cb + j * 16 + lm;
#pragma unroll
      for (int r = 0; r < 8; ++r) {
        const int n = rb + a * 16 + (l16 ? (r + 8) : r);
        ob[(size_t)n * NN + mcol] = acc[a][j][r];
      }
    }
  }
}

// Tuple element 0: copy xz through.
__global__ void __launch_bounds__(256) copy_xz_kernel(
    const float* __restrict__ xz, float* __restrict__ out) {
  const int i = blockIdx.x * 256 + threadIdx.x;
  if (i < MM) out[i] = xz[i];
}

extern "C" void kernel_launch(void* const* d_in, const int* in_sizes, int n_in,
                              void* d_out, int out_size, void* d_ws, size_t ws_size,
                              hipStream_t stream) {
  const float* xz        = (const float*)d_in[0];  // [M,1]
  const float* x         = (const float*)d_in[1];  // [B,N,1]
  const float* z         = (const float*)d_in[2];  // [B,C,M,K]
  const float* log_scale = (const float*)d_in[3];  // [1]

  float*    out = (float*)d_out;                   // 4096 xz floats + B*C*N*N
  _Float16* zt  = (_Float16*)d_ws;                 // 2 MB f16 staging

  // 1) zt = w^T @ z (f16 WMMA, f32 accumulate)
  zt_wmma_kernel<<<BB * CC * (NN / 128), 256, 0, stream>>>(xz, x, z, log_scale, zt);

  // 2) out = zt @ zt^T (one WMMA per 16x16 tile, K == 32)
  dim3 grid2(NN / 128, NN / 128, BB * CC);
  gram_wmma_kernel<<<grid2, 256, 0, stream>>>(zt, out + MM);

  // 3) tuple element 0
  copy_xz_kernel<<<(MM + 255) / 256, 256, 0, stream>>>(xz, out);
}
